// CausalMultiHeadAttention_22273700397173
// MI455X (gfx1250) — compile-verified, matrix operand order checked by reading, not run on hardware
//
#include <hip/hip_runtime.h>
#include <hip/hip_bf16.h>

typedef __bf16 bf16x16 __attribute__((ext_vector_type(16)));
typedef float  f32x8   __attribute__((ext_vector_type(8)));
typedef unsigned int u32x4 __attribute__((ext_vector_type(4)));

// exact operand type for the async-LDS builtin (per hipcc diagnostic)
typedef __attribute__((__vector_size__(4 * sizeof(int)))) int v4i;
typedef __attribute__((address_space(1))) v4i* gptr_v4i;
typedef __attribute__((address_space(3))) v4i* lptr_v4i;

#define D_MODEL 1024
#define N_HEADS 16
#define D_HEAD  64
#define BATCH   2
#define SEQ     2048
#define M_TOK   (BATCH * SEQ)   // 4096 tokens

// ---- CDNA5 async global->LDS path (guarded; fallback = load + ds_store) ----
#if defined(__has_builtin)
# if __has_builtin(__builtin_amdgcn_global_load_async_to_lds_b128) && \
     __has_builtin(__builtin_amdgcn_s_wait_asynccnt)
#  define USE_ASYNC_LDS 1
# endif
#endif
#ifndef USE_ASYNC_LDS
# define USE_ASYNC_LDS 0
#endif

__device__ __forceinline__ void stage16(const unsigned short* g, unsigned short* l) {
#if USE_ASYNC_LDS
    __builtin_amdgcn_global_load_async_to_lds_b128(
        (gptr_v4i)(unsigned short*)g, (lptr_v4i)l, 0, 0);
#else
    *(u32x4*)l = *(const u32x4*)g;
#endif
}

__device__ __forceinline__ unsigned short f2bf(float f) {
    unsigned int u = __float_as_uint(f);
    u += 0x7FFFu + ((u >> 16) & 1u);      // round-to-nearest-even
    return (unsigned short)(u >> 16);
}

// Load one 16x32 (A-style) / 32x16 (B-style) bf16 WMMA fragment.
// p must already include:  row(lane&15)*ld + (lane>>4)*8  element offset.
// Elements 0..7 from p[0..7] (K = half*8..), elements 8..15 from p[16..23].
__device__ __forceinline__ bf16x16 ldfrag(const unsigned short* p) {
    union { bf16x16 v; u32x4 q[2]; } u;
    u.q[0] = *(const u32x4*)p;
    u.q[1] = *(const u32x4*)(p + 16);
    return u.v;
}

__global__ void cvt_f32_bf16(const float* __restrict__ in,
                             unsigned short* __restrict__ out, int n) {
    int i = blockIdx.x * blockDim.x + threadIdx.x;
    if (i < n) out[i] = f2bf(in[i]);
}

// ---------------------------------------------------------------------------
// GEMM: out[m,n] = sum_k A[m,k] * W[n,k] + bias[n]   (K = N = 1024)
// LDS-staged, double-buffered (async global->LDS when available).
// Block = 256 threads (8 waves); block tile 128x128; wave tile 32x64.
// Per 32-wide K chunk the block stages A(128x32) + B(128x32) = 16 KB once.
// mode 0: bf16 -> [B,H,S,64] (Q,K)   mode 1: bf16 -> [B,H,64,S] (V^T)
// mode 2: fp32 -> [M_TOK,1024] (final output)
// ---------------------------------------------------------------------------
#define LDP 40   // padded LDS row stride (elements): bank-conflict-free b128 reads

__global__ __launch_bounds__(256) void gemm_bf16(
    const unsigned short* __restrict__ A,
    const unsigned short* __restrict__ W,
    const float* __restrict__ bias,
    void* __restrict__ outp,
    int mode)
{
    const int K = D_MODEL;
    __shared__ unsigned short ldsA[2][128 * LDP];
    __shared__ unsigned short ldsB[2][128 * LDP];

    int tid  = threadIdx.x;
    int lane = tid & 31;
    int wave = tid >> 5;
    int half = lane >> 4;
    int r    = lane & 15;
    int rg   = wave & 3;          // row group: rows rg*32 .. rg*32+31
    int cg   = wave >> 2;         // col group: cols cg*64 .. cg*64+63
    int mBase = blockIdx.x * 128;
    int nBase = blockIdx.y * 128;
    int m0 = mBase + rg * 32;
    int n0 = nBase + cg * 64;

    // staging assignment: 512 16-byte chunks per tile, 2 per thread
    int c0 = tid, c1 = tid + 256;
    int row0 = c0 >> 2, part0 = c0 & 3;
    int row1 = c1 >> 2, part1 = c1 & 3;

    f32x8 acc[2][4];
#pragma unroll
    for (int g = 0; g < 2; ++g)
#pragma unroll
        for (int u = 0; u < 4; ++u) acc[g][u] = 0.0f;

    auto stage = [&](int buf, int k0) {
        stage16(A + (size_t)(mBase + row0) * K + k0 + part0 * 8,
                &ldsA[buf][row0 * LDP + part0 * 8]);
        stage16(A + (size_t)(mBase + row1) * K + k0 + part1 * 8,
                &ldsA[buf][row1 * LDP + part1 * 8]);
        stage16(W + (size_t)(nBase + row0) * K + k0 + part0 * 8,
                &ldsB[buf][row0 * LDP + part0 * 8]);
        stage16(W + (size_t)(nBase + row1) * K + k0 + part1 * 8,
                &ldsB[buf][row1 * LDP + part1 * 8]);
    };

    stage(0, 0);
    const int NCHUNK = K / 32;    // 32
    for (int c = 0; c < NCHUNK; ++c) {
        int buf = c & 1;
        if (c + 1 < NCHUNK) {
            stage(buf ^ 1, (c + 1) * 32);
#if USE_ASYNC_LDS
            __builtin_amdgcn_s_wait_asynccnt(4);   // oldest 4 (this buffer) done
#endif
        } else {
#if USE_ASYNC_LDS
            __builtin_amdgcn_s_wait_asynccnt(0);
#endif
        }
        __syncthreads();

        bf16x16 af[2];
#pragma unroll
        for (int g = 0; g < 2; ++g)
            af[g] = ldfrag(&ldsA[buf][(rg * 32 + g * 16 + r) * LDP + half * 8]);
#pragma unroll
        for (int u = 0; u < 4; ++u) {
            bf16x16 bf_ = ldfrag(&ldsB[buf][(cg * 64 + 16 * u + r) * LDP + half * 8]);
#pragma unroll
            for (int g = 0; g < 2; ++g)
                acc[g][u] = __builtin_amdgcn_wmma_f32_16x16x32_bf16(
                    false, af[g], false, bf_, (short)0, acc[g][u], false, false);
        }
        __syncthreads();   // all waves done reading before buffer is overwritten
    }

    if (mode == 0) {
        unsigned short* out = (unsigned short*)outp;
#pragma unroll
        for (int g = 0; g < 2; ++g)
#pragma unroll
            for (int u = 0; u < 4; ++u) {
                int n = n0 + 16 * u + r;
                float bn = bias[n];
                int h = n >> 6, dh = n & 63;
#pragma unroll
                for (int j = 0; j < 8; ++j) {
                    int m = m0 + g * 16 + j + 8 * half;
                    int b = m >> 11, s = m & (SEQ - 1);
                    out[(((size_t)(b * N_HEADS + h)) * SEQ + s) * D_HEAD + dh] =
                        f2bf(acc[g][u][j] + bn);
                }
            }
    } else if (mode == 1) {
        unsigned short* out = (unsigned short*)outp;
#pragma unroll
        for (int g = 0; g < 2; ++g) {
            int mb = m0 + g * 16 + 8 * half;
            int b = mb >> 11, s = mb & (SEQ - 1);
#pragma unroll
            for (int u = 0; u < 4; ++u) {
                int n = n0 + 16 * u + r;
                float bn = bias[n];
                int h = n >> 6, dh = n & 63;
                union { unsigned short tmp[8]; u32x4 v; } uu;
#pragma unroll
                for (int j = 0; j < 8; ++j) uu.tmp[j] = f2bf(acc[g][u][j] + bn);
                *(u32x4*)(out + (((size_t)(b * N_HEADS + h)) * D_HEAD + dh) * SEQ + s) = uu.v;
            }
        }
    } else {
        float* out = (float*)outp;
#pragma unroll
        for (int g = 0; g < 2; ++g)
#pragma unroll
            for (int u = 0; u < 4; ++u) {
                int n = n0 + 16 * u + r;
                float bn = bias[n];
#pragma unroll
                for (int j = 0; j < 8; ++j)
                    out[(size_t)(m0 + g * 16 + j + 8 * half) * D_MODEL + n] =
                        acc[g][u][j] + bn;
            }
    }
}

// ---------------------------------------------------------------------------
// Flash attention, causal.  Q,K: [B,H,S,64] bf16; Vt: [B,H,64,S] bf16.
// attn_out: [B,S,1024] bf16.  block = 128 threads (4 waves), 16 q-rows/wave.
// ---------------------------------------------------------------------------
__global__ __launch_bounds__(128) void attn_flash(
    const unsigned short* __restrict__ Q,
    const unsigned short* __restrict__ Kk,
    const unsigned short* __restrict__ Vt,
    unsigned short* __restrict__ attn_out)
{
    int bh   = blockIdx.y;              // b*16 + h
    int qb   = blockIdx.x;              // 64-row query block
    int lane = threadIdx.x & 31;
    int wave = threadIdx.x >> 5;
    int half = lane >> 4;
    int r    = lane & 15;
    int q0   = qb * 64 + wave * 16;

    __shared__ unsigned short ldsP[4][16 * 64];   // private 16x64 P tile per wave
    unsigned short* myP = &ldsP[wave][0];

    const unsigned short* qp = Q + ((size_t)bh * SEQ + q0 + r) * D_HEAD + half * 8;
    bf16x16 qf0 = ldfrag(qp);
    bf16x16 qf1 = ldfrag(qp + 32);

    f32x8 accO[4];
#pragma unroll
    for (int u = 0; u < 4; ++u) accO[u] = 0.0f;
    float mrow[8], lrow[8];
#pragma unroll
    for (int j = 0; j < 8; ++j) { mrow[j] = -__builtin_inff(); lrow[j] = 0.0f; }

    const float scale = 0.125f;   // 1/sqrt(64)

    for (int kb = 0; kb <= qb; ++kb) {
        int key0 = kb * 64;
        bool diag = (kb == qb);

        if (kb + 1 <= qb) {  // prefetch next key block (K rows + V^T rows)
            __builtin_prefetch(Kk + ((size_t)bh * SEQ + key0 + 64 + lane * 2) * D_HEAD, 0, 1);
            __builtin_prefetch(Vt + ((size_t)bh * D_HEAD + lane * 2) * SEQ + key0 + 64, 0, 1);
        }

        // ---- scores S = Q * K^T  (16 x 64) ----
        f32x8 sc[4];
#pragma unroll
        for (int t = 0; t < 4; ++t) sc[t] = 0.0f;
#pragma unroll
        for (int t = 0; t < 4; ++t) {
            const unsigned short* kp =
                Kk + ((size_t)bh * SEQ + key0 + 16 * t + r) * D_HEAD + half * 8;
            bf16x16 kf0 = ldfrag(kp);
            bf16x16 kf1 = ldfrag(kp + 32);
            sc[t] = __builtin_amdgcn_wmma_f32_16x16x32_bf16(
                false, qf0, false, kf0, (short)0, sc[t], false, false);
            sc[t] = __builtin_amdgcn_wmma_f32_16x16x32_bf16(
                false, qf1, false, kf1, (short)0, sc[t], false, false);
        }

        // ---- scale + causal mask + online softmax ----
#pragma unroll
        for (int j = 0; j < 8; ++j) {
            int row = q0 + j + 8 * half;
            float mx = -__builtin_inff();
#pragma unroll
            for (int t = 0; t < 4; ++t) {
                float s = sc[t][j] * scale;
                if (diag && (key0 + 16 * t + r) > row) s = -__builtin_inff();
                sc[t][j] = s;
                mx = fmaxf(mx, s);
            }
#pragma unroll
            for (int off = 1; off < 16; off <<= 1)
                mx = fmaxf(mx, __shfl_xor(mx, off, 16));
            float mnew  = fmaxf(mrow[j], mx);
            float alpha = __expf(mrow[j] - mnew);
            mrow[j] = mnew;
            float rs = 0.0f;
#pragma unroll
            for (int t = 0; t < 4; ++t) {
                float p = __expf(sc[t][j] - mnew);
                sc[t][j] = p;
                rs += p;
            }
#pragma unroll
            for (int off = 1; off < 16; off <<= 1)
                rs += __shfl_xor(rs, off, 16);
            lrow[j] = lrow[j] * alpha + rs;
#pragma unroll
            for (int u = 0; u < 4; ++u) accO[u][j] *= alpha;
        }

        // ---- C-layout -> A-layout via private LDS strip ----
#pragma unroll
        for (int t = 0; t < 4; ++t)
#pragma unroll
            for (int j = 0; j < 8; ++j)
                myP[(j + 8 * half) * 64 + 16 * t + r] = f2bf(sc[t][j]);

        // ---- O += P * V ----
#pragma unroll
        for (int c = 0; c < 2; ++c) {
            const unsigned short* pp = myP + r * 64 + c * 32 + half * 8;
            union { bf16x16 v; u32x4 q[2]; } up;
            up.q[0] = *(const u32x4*)pp;
            up.q[1] = *(const u32x4*)(pp + 16);
            bf16x16 pf = up.v;
#pragma unroll
            for (int u = 0; u < 4; ++u) {
                const unsigned short* vp =
                    Vt + ((size_t)bh * D_HEAD + 16 * u + r) * SEQ
                       + key0 + c * 32 + half * 8;
                bf16x16 vf = ldfrag(vp);
                accO[u] = __builtin_amdgcn_wmma_f32_16x16x32_bf16(
                    false, pf, false, vf, (short)0, accO[u], false, false);
            }
        }
    }

    // ---- normalize + store to [B,S,1024] ----
    int b = bh >> 4, h = bh & 15;
#pragma unroll
    for (int u = 0; u < 4; ++u) {
        int dh = 16 * u + r;
#pragma unroll
        for (int j = 0; j < 8; ++j) {
            int row = q0 + j + 8 * half;
            float o = accO[u][j] / lrow[j];
            attn_out[((size_t)b * SEQ + row) * D_MODEL + h * D_HEAD + dh] = f2bf(o);
        }
    }
}

// ---------------------------------------------------------------------------
extern "C" void kernel_launch(void* const* d_in, const int* in_sizes, int n_in,
                              void* d_out, int out_size, void* d_ws, size_t ws_size,
                              hipStream_t stream) {
    const float* x  = (const float*)d_in[0];
    // d_in[1] = mask (causality hard-coded)
    const float* Wq = (const float*)d_in[2];
    const float* bq = (const float*)d_in[3];
    const float* Wk = (const float*)d_in[4];
    const float* bk = (const float*)d_in[5];
    const float* Wv = (const float*)d_in[6];
    const float* bv = (const float*)d_in[7];
    const float* Wo = (const float*)d_in[8];
    const float* bo = (const float*)d_in[9];
    float* out = (float*)d_out;

    const size_t SZ_X  = (size_t)M_TOK * D_MODEL * 2;   // 8 MB
    const size_t SZ_W  = (size_t)D_MODEL * D_MODEL * 2; // 2 MB
    const size_t SZ_T  = (size_t)M_TOK * D_MODEL * 2;   // 8 MB
    char* ws = (char*)d_ws;
    unsigned short* xb    = (unsigned short*)(ws);
    unsigned short* wqb   = (unsigned short*)(ws + SZ_X);
    unsigned short* wkb   = (unsigned short*)(ws + SZ_X + SZ_W);
    unsigned short* wvb   = (unsigned short*)(ws + SZ_X + 2 * SZ_W);
    unsigned short* wob   = (unsigned short*)(ws + SZ_X + 3 * SZ_W);
    unsigned short* Qb    = (unsigned short*)(ws + SZ_X + 4 * SZ_W);
    unsigned short* Kb    = (unsigned short*)(ws + SZ_X + 4 * SZ_W + SZ_T);
    unsigned short* Vtb   = (unsigned short*)(ws + SZ_X + 4 * SZ_W + 2 * SZ_T);
    unsigned short* attnb = (unsigned short*)(ws + SZ_X + 4 * SZ_W + 3 * SZ_T);
    if (ws_size < SZ_X + 4 * SZ_W + 4 * SZ_T) return;

    {
        int nx = M_TOK * D_MODEL;
        cvt_f32_bf16<<<(nx + 255) / 256, 256, 0, stream>>>(x, xb, nx);
        int nw = D_MODEL * D_MODEL;
        cvt_f32_bf16<<<(nw + 255) / 256, 256, 0, stream>>>(Wq, wqb, nw);
        cvt_f32_bf16<<<(nw + 255) / 256, 256, 0, stream>>>(Wk, wkb, nw);
        cvt_f32_bf16<<<(nw + 255) / 256, 256, 0, stream>>>(Wv, wvb, nw);
        cvt_f32_bf16<<<(nw + 255) / 256, 256, 0, stream>>>(Wo, wob, nw);
    }

    dim3 gg(M_TOK / 128, D_MODEL / 128);   // (32, 8)
    gemm_bf16<<<gg, 256, 0, stream>>>(xb, wqb, bq, (void*)Qb, 0);
    gemm_bf16<<<gg, 256, 0, stream>>>(xb, wkb, bk, (void*)Kb, 0);
    gemm_bf16<<<gg, 256, 0, stream>>>(xb, wvb, bv, (void*)Vtb, 1);

    dim3 ga(SEQ / 64, BATCH * N_HEADS);    // (32, 32)
    attn_flash<<<ga, 128, 0, stream>>>(Qb, Kb, Vtb, attnb);

    gemm_bf16<<<gg, 256, 0, stream>>>(attnb, wob, bo, (void*)out, 2);
}
